// TDGNNModel_32547262169237
// MI455X (gfx1250) — matrix operand order repaired and checked by reading, not compiled, not run on hardware
//
#include <hip/hip_runtime.h>
#include <hip/hip_bf16.h>
#include <math.h>

typedef __attribute__((ext_vector_type(16))) _Float16 v16h;
typedef __attribute__((ext_vector_type(8)))  _Float16 v8h;
typedef __attribute__((ext_vector_type(8)))  float    v8f;

#define N_NODES 20000
#define N_EDGES 50000
#define HID     128
#define OUT_DIM 64
#define N_PAIRS 128
#define N_U     256     // flattened targets
#define CAP     512     // max incident edges tracked per target

// ---------------------------------------------------------------------------
// f32 -> f16 convert
// ---------------------------------------------------------------------------
__global__ void cvt_f32_f16(const float* __restrict__ in, _Float16* __restrict__ out, int n) {
    int i = blockIdx.x * blockDim.x + threadIdx.x;
    if (i < n) out[i] = (_Float16)in[i];
}

// ---------------------------------------------------------------------------
// WMMA GEMM: C[M,N] = act(A[M,128] @ W[N,128]^T + bias), f16 in, f32 acc.
// N-register-blocked: each wave owns a 16-row strip and sweeps all NT=N/16
// column tiles, loading its A fragment ONCE per k-step (A traffic /NT).
// Per k-step all NT B fragments are staged into registers first so the
// scheduler can clause the loads and run the NT WMMAs back-to-back.
// Block = 256 threads = 8 waves = 128 rows. K fixed at 128.
// ---------------------------------------------------------------------------
template <int NT>
__global__ void gemm_k128(const _Float16* __restrict__ A,
                          const _Float16* __restrict__ W,
                          const float* __restrict__ bias,
                          float* __restrict__ Cf,
                          _Float16* __restrict__ Ch,
                          int M, int relu)
{
    constexpr int N = NT * 16;
    const int wave = threadIdx.x >> 5;
    const int lane = threadIdx.x & 31;
    const int m0   = (blockIdx.x * 8 + wave) * 16;
    if (m0 >= M) return;                      // wave-uniform: EXEC stays all-ones

    const int hs = lane >> 4;                 // 0: lanes 0-15, 1: lanes 16-31
    const int r  = lane & 15;

    // A fragment source: row m0+r, K chunks {kb..kb+7, kb+16..kb+23}, kb = 8*hs
    const _Float16* arow = A + (size_t)(m0 + r) * 128 + hs * 8;
    // B fragment source: weight row (nt*16 + r), K chunks {kb..kb+7, kb+16..kb+23}, kb = 8*hs
    const _Float16* wrow = W + (size_t)r * 128 + hs * 8;

    __builtin_prefetch(arow, 0, 3);           // global_prefetch_b8

    v8f acc[NT];
#pragma unroll
    for (int nt = 0; nt < NT; ++nt) acc[nt] = (v8f){};

#pragma unroll
    for (int k0 = 0; k0 < 128; k0 += 32) {
        v8h a_lo = *reinterpret_cast<const v8h*>(arow + k0);
        v8h a_hi = *reinterpret_cast<const v8h*>(arow + k0 + 16);
        v16h a;
#pragma unroll
        for (int i = 0; i < 8; ++i) { a[i] = a_lo[i]; a[i + 8] = a_hi[i]; }

        // stage all B fragments for this k-step (clause-able load burst)
        v16h b[NT];
#pragma unroll
        for (int nt = 0; nt < NT; ++nt) {
            const _Float16* wp = wrow + (size_t)nt * 16 * 128 + k0;
            v8h b_lo = *reinterpret_cast<const v8h*>(wp);
            v8h b_hi = *reinterpret_cast<const v8h*>(wp + 16);
#pragma unroll
            for (int i = 0; i < 8; ++i) { b[nt][i] = b_lo[i]; b[nt][i + 8] = b_hi[i]; }
        }
        // back-to-back WMMA chain (independent accumulators, no RAW hazards)
#pragma unroll
        for (int nt = 0; nt < NT; ++nt)
            acc[nt] = __builtin_amdgcn_wmma_f32_16x16x32_f16(
                          false, a, false, b[nt], (short)0, acc[nt], false, false);
    }

    const int mbase = m0 + hs * 8;
#pragma unroll
    for (int nt = 0; nt < NT; ++nt) {
        const int n    = nt * 16 + r;
        const float bv = bias ? bias[n] : 0.0f;
#pragma unroll
        for (int v = 0; v < 8; ++v) {
            float c = acc[nt][v] + bv;
            if (relu) c = fmaxf(c, 0.0f);
            size_t off = (size_t)(mbase + v) * N + n;
            if (Cf) Cf[off] = c;
            if (Ch) Ch[off] = (_Float16)c;
        }
    }
}

static void launch_gemm(const _Float16* A, const _Float16* W, const float* bias,
                        float* Cf, _Float16* Ch, int M, int N, int relu,
                        hipStream_t s)
{
    int mtiles = (M + 15) / 16;
    dim3 grid((mtiles + 7) / 8);
    if (N == 128)
        gemm_k128<8><<<grid, 256, 0, s>>>(A, W, bias, Cf, Ch, M, relu);
    else // N == 64
        gemm_k128<4><<<grid, 256, 0, s>>>(A, W, bias, Cf, Ch, M, relu);
}

// ---------------------------------------------------------------------------
// time encoder stage 1: t1[e,j] = relu(ts[e]*W1[j] + b1[j])  (f16 out)
// ---------------------------------------------------------------------------
__global__ void time_encode(const float* __restrict__ ts, const float* __restrict__ W1,
                            const float* __restrict__ b1, _Float16* __restrict__ t1, int E)
{
    int i = blockIdx.x * blockDim.x + threadIdx.x;
    if (i >= E * 128) return;
    int e = i >> 7, j = i & 127;
    float v = ts[e] * W1[j] + b1[j];
    t1[i] = (_Float16)fmaxf(v, 0.0f);
}

// ---------------------------------------------------------------------------
// row gather (f16), rows*128 elements
// ---------------------------------------------------------------------------
__global__ void gather_rows_h(const _Float16* __restrict__ src, const int* __restrict__ idx,
                              _Float16* __restrict__ dst, int rows)
{
    int i = blockIdx.x * blockDim.x + threadIdx.x;
    if (i >= rows * 128) return;
    dst[i] = src[(size_t)idx[i >> 7] * 128 + (i & 127)];
}

// ---------------------------------------------------------------------------
// deterministic incident-edge list builder: one block per target,
// two-pass strided scan + serial prefix (order stable across replays)
// ---------------------------------------------------------------------------
__global__ void build_lists(const int* __restrict__ src, const int* __restrict__ dst,
                            const int* __restrict__ targets,
                            int* __restrict__ lists, int* __restrict__ counts, int E)
{
    __shared__ int cnts[256];
    __shared__ int offs[256];
    const int u = blockIdx.x, tid = threadIdx.x;
    const int t = targets[u];
    int c = 0;
    for (int e = tid; e < E; e += 256)
        if (src[e] == t || dst[e] == t) c++;
    cnts[tid] = c;
    __syncthreads();
    if (tid == 0) {
        int acc = 0;
        for (int i = 0; i < 256; ++i) { offs[i] = acc; acc += cnts[i]; }
        counts[u] = (acc < CAP) ? acc : CAP;
    }
    __syncthreads();
    int pos = offs[tid];
    for (int e = tid; e < E; e += 256) {
        if (src[e] == t || dst[e] == t) {
            if (pos < CAP) lists[u * CAP + pos] = e;
            pos++;
        }
    }
}

// ---------------------------------------------------------------------------
// sparse multi-head attention over incident edges; one block (128 thr) per u
// ctx[u] = sum_e softmax(scale * q.(Ak[nbr]+Bk[e]))_e * (Av[nbr]+Bv[e])
// ---------------------------------------------------------------------------
__global__ void attention_kernel(const float* __restrict__ q,
                                 const _Float16* __restrict__ Akh, const _Float16* __restrict__ Avh,
                                 const _Float16* __restrict__ Bkh, const _Float16* __restrict__ Bvh,
                                 const int* __restrict__ src, const int* __restrict__ dst,
                                 const int* __restrict__ targets,
                                 const int* __restrict__ lists, const int* __restrict__ counts,
                                 _Float16* __restrict__ ctx_h, int* __restrict__ hasnbr,
                                 float scale)
{
    __shared__ float sc[4][CAP];
    __shared__ int   nbrs[CAP];
    __shared__ float qs[128];
    const int u  = blockIdx.x;
    const int t  = threadIdx.x;        // 128 threads
    const int tg = targets[u];
    const int L  = counts[u];
    qs[t] = q[u * 128 + t];
    __syncthreads();

    for (int i = t; i < L; i += 128) {
        int e = lists[u * CAP + i];
        int s = src[e], d = dst[e];
        int nb = (s == tg) ? d : s;
        nbrs[i] = nb;
        const _Float16* ak = Akh + (size_t)nb * 128;
        const _Float16* bk = Bkh + (size_t)e * 128;
        for (int h = 0; h < 4; ++h) {
            float dot = 0.0f;
            for (int k = 0; k < 32; ++k)
                dot += qs[h * 32 + k] * ((float)ak[h * 32 + k] + (float)bk[h * 32 + k]);
            sc[h][i] = dot * scale;
        }
    }
    __syncthreads();

    if (t < 4 && L > 0) {              // per-head streaming softmax (serial, L is tiny)
        float m = -1e30f;
        for (int i = 0; i < L; ++i) m = fmaxf(m, sc[t][i]);
        float z = 0.0f;
        for (int i = 0; i < L; ++i) { float pe = expf(sc[t][i] - m); sc[t][i] = pe; z += pe; }
        float inv = 1.0f / z;
        for (int i = 0; i < L; ++i) sc[t][i] *= inv;
    }
    __syncthreads();

    const int h = t >> 5, k = t & 31;
    float acc = 0.0f;
    for (int i = 0; i < L; ++i) {
        int e = lists[u * CAP + i];
        int nb = nbrs[i];
        acc += sc[h][i] * ((float)Avh[(size_t)nb * 128 + h * 32 + k] +
                           (float)Bvh[(size_t)e  * 128 + h * 32 + k]);
    }
    ctx_h[u * 128 + t] = (_Float16)acc;
    if (t == 0) hasnbr[u] = (L > 0) ? 1 : 0;
}

// ---------------------------------------------------------------------------
// agg_in = has_nbr ? attn : tf[target]
// ---------------------------------------------------------------------------
__global__ void select_aggin(const _Float16* __restrict__ attn_h, const _Float16* __restrict__ tf_h,
                             const int* __restrict__ targets, const int* __restrict__ hasnbr,
                             _Float16* __restrict__ aggin)
{
    int i = blockIdx.x * blockDim.x + threadIdx.x;
    if (i >= N_U * 128) return;
    int u = i >> 7;
    aggin[i] = hasnbr[u] ? attn_h[i] : tf_h[(size_t)targets[u] * 128 + (i & 127)];
}

// ---------------------------------------------------------------------------
// x[targets[u], :] = agg[u, :]   (duplicate targets write identical rows)
// ---------------------------------------------------------------------------
__global__ void scatter_rows(const float* __restrict__ agg, const int* __restrict__ targets,
                             float* __restrict__ x)
{
    int i = blockIdx.x * blockDim.x + threadIdx.x;
    if (i >= N_U * 128) return;
    x[(size_t)targets[i >> 7] * 128 + (i & 127)] = agg[i];
}

// ---------------------------------------------------------------------------
// x = relu(x); xh = (f16)x
// ---------------------------------------------------------------------------
__global__ void relu_cvt(float* __restrict__ x, _Float16* __restrict__ xh, int n)
{
    int i = blockIdx.x * blockDim.x + threadIdx.x;
    if (i >= n) return;
    float v = fmaxf(x[i], 0.0f);
    x[i] = v;
    xh[i] = (_Float16)v;
}

// ---------------------------------------------------------------------------
// link-prediction MLP head (f32, matches reference output precision)
// pair p: concat(emb[2p], emb[2p+1]) -> 128 -> 64 -> sigmoid
// ---------------------------------------------------------------------------
__global__ void pair_mlp(const float* __restrict__ emb,
                         const float* __restrict__ W1, const float* __restrict__ b1,
                         const float* __restrict__ W2, const float* __restrict__ b2,
                         const float* __restrict__ W3, const float* __restrict__ b3,
                         float* __restrict__ out)
{
    __shared__ float pv[128], z1[128], z2[64], red[64];
    const int p = blockIdx.x, t = threadIdx.x;     // 128 threads
    pv[t] = (t < 64) ? emb[(size_t)(2 * p) * 64 + t]
                     : emb[(size_t)(2 * p + 1) * 64 + (t - 64)];
    __syncthreads();
    float a = b1[t];
    for (int k = 0; k < 128; ++k) a += W1[t * 128 + k] * pv[k];
    z1[t] = fmaxf(a, 0.0f);
    __syncthreads();
    if (t < 64) {
        float a2 = b2[t];
        for (int k = 0; k < 128; ++k) a2 += W2[t * 128 + k] * z1[k];
        z2[t] = fmaxf(a2, 0.0f);
    }
    __syncthreads();
    if (t < 64) red[t] = W3[t] * z2[t];
    __syncthreads();
    if (t == 0) {
        float s = b3[0];
        for (int k = 0; k < 64; ++k) s += red[k];
        out[p] = 1.0f / (1.0f + expf(-s));
    }
}

// ===========================================================================
extern "C" void kernel_launch(void* const* d_in, const int* in_sizes, int n_in,
                              void* d_out, int out_size, void* d_ws, size_t ws_size,
                              hipStream_t stream)
{
    (void)in_sizes; (void)n_in; (void)out_size; (void)ws_size;

    const float* node_features = (const float*)d_in[0];
    const int*   edge_index    = (const int*)  d_in[1];
    const float* edge_ts       = (const float*)d_in[2];
    const int*   targets       = (const int*)  d_in[3];   // target_pairs flattened [256]
    const float* in_W   = (const float*)d_in[4];  const float* in_b   = (const float*)d_in[5];
    const float* te_W1  = (const float*)d_in[6];  const float* te_b1  = (const float*)d_in[7];
    const float* te_W2  = (const float*)d_in[8];  const float* te_b2  = (const float*)d_in[9];
    const float* ai_W   = (const float*)d_in[10]; const float* ai_b   = (const float*)d_in[11];
    const float* ao_W   = (const float*)d_in[12]; const float* ao_b   = (const float*)d_in[13];
    const float* ft_W   = (const float*)d_in[14]; const float* ft_b   = (const float*)d_in[15];
    const float* op_W   = (const float*)d_in[16]; const float* op_b   = (const float*)d_in[17];
    const float* o1_W   = (const float*)d_in[18]; const float* o1_b   = (const float*)d_in[19];
    const float* o2_W   = (const float*)d_in[20]; const float* o2_b   = (const float*)d_in[21];
    const float* lp1_W  = (const float*)d_in[22]; const float* lp1_b  = (const float*)d_in[23];
    const float* lp2_W  = (const float*)d_in[24]; const float* lp2_b  = (const float*)d_in[25];
    const float* lp3_W  = (const float*)d_in[26]; const float* lp3_b  = (const float*)d_in[27];

    const int* src = edge_index;
    const int* dst = edge_index + N_EDGES;

    // ---- bump allocator over workspace ------------------------------------
    char* ws = (char*)d_ws;
    size_t off = 0;
    auto alloc = [&](size_t bytes) -> void* {
        void* p = ws + off;
        off = (off + bytes + 255) & ~(size_t)255;
        return p;
    };
    const size_t NH = (size_t)N_NODES * HID;   // 2.56M
    const size_t EH = (size_t)N_EDGES * HID;   // 6.4M

    _Float16* nf_h   = (_Float16*)alloc(NH * 2);
    float*    x_f    = (float*)   alloc(NH * 4);
    _Float16* x_h    = (_Float16*)alloc(NH * 2);
    _Float16* tf_h   = (_Float16*)alloc(NH * 2);
    _Float16* Ak_h   = (_Float16*)alloc(NH * 2);
    _Float16* Av_h   = (_Float16*)alloc(NH * 2);
    _Float16* t1_h   = (_Float16*)alloc(EH * 2);
    _Float16* tfe_h  = (_Float16*)alloc(EH * 2);
    _Float16* Bk_h   = (_Float16*)alloc(EH * 2);
    _Float16* Bv_h   = (_Float16*)alloc(EH * 2);
    float*    q_f    = (float*)   alloc(N_U * 128 * 4);
    _Float16* ctx_h  = (_Float16*)alloc(N_U * 128 * 2);
    _Float16* attn_h = (_Float16*)alloc(N_U * 128 * 2);
    _Float16* aggin_h= (_Float16*)alloc(N_U * 128 * 2);
    float*    agg_f  = (float*)   alloc(N_U * 128 * 4);
    _Float16* tg_h   = (_Float16*)alloc(N_U * 128 * 2);
    _Float16* y1_h   = (_Float16*)alloc(N_U * 128 * 2);
    float*    emb_f  = (float*)   alloc(N_U * OUT_DIM * 4);
    int*      lists  = (int*)     alloc((size_t)N_U * CAP * 4);
    int*      counts = (int*)     alloc(N_U * 4);
    int*      hasnbr = (int*)     alloc(N_U * 4);
    // f16 weight mirrors
    _Float16* inW_h  = (_Float16*)alloc(16384 * 2);
    _Float16* teW2_h = (_Float16*)alloc(32768 * 2);
    _Float16* aiW_h  = (_Float16*)alloc(98304 * 2);   // [2][3*128][128]
    _Float16* aoW_h  = (_Float16*)alloc(32768 * 2);
    _Float16* ftW_h  = (_Float16*)alloc(32768 * 2);
    _Float16* opW_h  = (_Float16*)alloc(32768 * 2);
    _Float16* o1W_h  = (_Float16*)alloc(16384 * 2);
    _Float16* o2W_h  = (_Float16*)alloc(8192 * 2);

    auto cvt = [&](const float* a, _Float16* b, int n) {
        cvt_f32_f16<<<(n + 255) / 256, 256, 0, stream>>>(a, b, n);
    };

    // ---- weight + feature conversion --------------------------------------
    cvt(node_features, nf_h, (int)NH);
    cvt(in_W, inW_h, 16384);
    cvt(te_W2, teW2_h, 32768);
    cvt(ai_W, aiW_h, 98304);
    cvt(ao_W, aoW_h, 32768);
    cvt(ft_W, ftW_h, 32768);
    cvt(op_W, opW_h, 32768);
    cvt(o1_W, o1W_h, 16384);
    cvt(o2_W, o2W_h, 8192);

    // ---- x = node_features @ in_W^T + in_b --------------------------------
    launch_gemm(nf_h, inW_h, in_b, x_f, x_h, N_NODES, HID, 0, stream);

    const float scale = 0.17677669529663687f;   // 1/sqrt(32)

    for (int l = 0; l < 2; ++l) {
        const _Float16* Wq = aiW_h + (size_t)l * 49152;
        const _Float16* Wk = Wq + 16384;
        const _Float16* Wv = Wq + 32768;
        const float* bq = ai_b + l * 384;
        const float* bk = bq + 128;
        const float* bv = bq + 256;

        // tf = x @ ft_W^T + ft_b
        launch_gemm(x_h, ftW_h + (size_t)l * 16384, ft_b + l * 128,
                    nullptr, tf_h, N_NODES, HID, 0, stream);
        // time encoder
        time_encode<<<(N_EDGES * 128 + 255) / 256, 256, 0, stream>>>(
            edge_ts, te_W1 + l * 128, te_b1 + l * 128, t1_h, N_EDGES);
        launch_gemm(t1_h, teW2_h + (size_t)l * 16384, te_b2 + l * 128,
                    nullptr, tfe_h, N_EDGES, HID, 0, stream);
        // q = tf[targets] @ Wq^T + bq
        gather_rows_h<<<(N_U * 128 + 255) / 256, 256, 0, stream>>>(tf_h, targets, tg_h, N_U);
        launch_gemm(tg_h, Wq, bq, q_f, nullptr, N_U, HID, 0, stream);
        // node / edge key & value terms
        launch_gemm(tf_h,  Wk, nullptr, nullptr, Ak_h, N_NODES, HID, 0, stream);
        launch_gemm(tf_h,  Wv, nullptr, nullptr, Av_h, N_NODES, HID, 0, stream);
        launch_gemm(tfe_h, Wk, bk,      nullptr, Bk_h, N_EDGES, HID, 0, stream);
        launch_gemm(tfe_h, Wv, bv,      nullptr, Bv_h, N_EDGES, HID, 0, stream);
        // sparse attention
        build_lists<<<N_U, 256, 0, stream>>>(src, dst, targets, lists, counts, N_EDGES);
        attention_kernel<<<N_U, 128, 0, stream>>>(q_f, Ak_h, Av_h, Bk_h, Bv_h,
                                                  src, dst, targets, lists, counts,
                                                  ctx_h, hasnbr, scale);
        // attn projection, gated aggregation, op projection
        launch_gemm(ctx_h, aoW_h + (size_t)l * 16384, ao_b + l * 128,
                    nullptr, attn_h, N_U, HID, 0, stream);
        select_aggin<<<(N_U * 128 + 255) / 256, 256, 0, stream>>>(
            attn_h, tf_h, targets, hasnbr, aggin_h);
        launch_gemm(aggin_h, opW_h + (size_t)l * 16384, op_b + l * 128,
                    agg_f, nullptr, N_U, HID, 0, stream);
        // x.at[targets].set(agg); x = relu(x)
        scatter_rows<<<(N_U * 128 + 255) / 256, 256, 0, stream>>>(agg_f, targets, x_f);
        relu_cvt<<<((int)NH + 255) / 256, 256, 0, stream>>>(x_f, x_h, (int)NH);
    }

    // ---- output head: only target rows are consumed downstream ------------
    gather_rows_h<<<(N_U * 128 + 255) / 256, 256, 0, stream>>>(x_h, targets, tg_h, N_U);
    launch_gemm(tg_h, o1W_h, o1_b, nullptr, y1_h, N_U, HID, 1, stream);
    launch_gemm(y1_h, o2W_h, o2_b, emb_f, nullptr, N_U, OUT_DIM, 0, stream);

    // ---- link prediction MLP ----------------------------------------------
    pair_mlp<<<N_PAIRS, 128, 0, stream>>>(emb_f, lp1_W, lp1_b, lp2_W, lp2_b,
                                          lp3_W, lp3_b, (float*)d_out);
}
